// CategoricalDuelNetwork1_15075335209624
// MI455X (gfx1250) — matrix operand order repaired and checked by reading, not compiled, not run on hardware
//
#include <hip/hip_runtime.h>
#include <hip/hip_bf16.h>
#include <math.h>

typedef __attribute__((ext_vector_type(16))) _Float16 v16h;
typedef __attribute__((ext_vector_type(8)))  _Float16 v8h;
typedef __attribute__((ext_vector_type(8)))  float    v8f;

#define BATCH   32768
#define STATE   272
#define DFLAT   484      // 11 * 44
#define KP      512      // 484 padded to mult of 32
#define XDIM    544      // 272+272, already mult of 32
#define NACT    30
#define ATOMS   51
#define QROW    1530     // 30*51

__constant__ int c_sizes[11] = {11,29,29,29,11,29,29,1,44,30,30};
__constant__ int c_offs [11] = {0,11,40,69,98,109,138,167,168,212,242};

// ---------------------------------------------------------------------------
// Weight conversion: f32 (N,K) [or transposed (K,N)] -> f16 [Npad][Kpad], 0-pad
// ---------------------------------------------------------------------------
__global__ __launch_bounds__(256) void convw_kernel(
    _Float16* __restrict__ dst, const float* __restrict__ src,
    int N, int K, int Npad, int Kpad, int trans)
{
    int idx = blockIdx.x * 256 + threadIdx.x;
    int total = Npad * Kpad;
    if (idx >= total) return;
    int n = idx / Kpad, k = idx % Kpad;
    float v = 0.f;
    if (n < N && k < K) v = trans ? src[(size_t)k * N + n] : src[(size_t)n * K + k];
    dst[idx] = (_Float16)v;
}

// ---------------------------------------------------------------------------
// Preprocess: wave-per-row. Computes att1 = (l2norm-attn @ padded1) for input1,
// plus padded flats of inputs / input2 as f16, plus zero tails.
// ---------------------------------------------------------------------------
__global__ __launch_bounds__(256) void preprocess_kernel(
    const float* __restrict__ in1, const float* __restrict__ in2,
    const float* __restrict__ ins,
    _Float16* __restrict__ att1h, _Float16* __restrict__ pad2h,
    _Float16* __restrict__ padsh, _Float16* __restrict__ pooledh)
{
    __shared__ float s_pad [8][DFLAT];
    __shared__ float s_rn  [8][16];
    __shared__ float s_attn[8][128];
    const int lane = threadIdx.x & 31;
    const int w    = threadIdx.x >> 5;
    const int b    = blockIdx.x * 8 + w;
    const float* x1 = in1 + (size_t)b * STATE;
    const size_t rb = (size_t)b * KP;

    for (int f = lane; f < DFLAT; f += 32) {
        int i = f / 44, pos = f % 44;
        bool valid = pos < c_sizes[i];
        int  src   = c_offs[i] + pos;
        float v1 = valid ? x1[src] : 0.f;
        s_pad[w][f] = v1;
        padsh[rb + f] = (_Float16)(valid ? ins[(size_t)b * STATE + src] : 0.f);
        pad2h[rb + f] = (_Float16)(valid ? in2[(size_t)b * STATE + src] : 0.f);
    }
    for (int f = DFLAT + lane; f < KP; f += 32) {
        att1h  [rb + f] = (_Float16)0.f;
        padsh  [rb + f] = (_Float16)0.f;
        pad2h  [rb + f] = (_Float16)0.f;
        pooledh[rb + f] = (_Float16)0.f;
    }
    __syncthreads();
    if (lane < 11) {
        float ss = 0.f;
        for (int j = 0; j < 44; ++j) { float v = s_pad[w][lane*44 + j]; ss += v*v; }
        s_rn[w][lane] = 1.f / fmaxf(sqrtf(ss), 1e-12f);
    }
    __syncthreads();
    for (int idx = lane; idx < 121; idx += 32) {
        int i = idx / 11, j = idx % 11;
        float d = 0.f;
        for (int t = 0; t < 44; ++t) d += s_pad[w][i*44 + t] * s_pad[w][j*44 + t];
        s_attn[w][idx] = d * s_rn[w][i] * s_rn[w][j];
    }
    __syncthreads();
    for (int f = lane; f < DFLAT; f += 32) {
        int i = f / 44, pos = f % 44;
        float a = 0.f;
        #pragma unroll
        for (int j = 0; j < 11; ++j) a += s_attn[w][i*11 + j] * s_pad[w][j*44 + pos];
        att1h[rb + f] = (_Float16)a;
    }
}

// ---------------------------------------------------------------------------
// Generic WMMA GEMM: C[m,n] = epi( sum_k A[m,k]*W[n,k] + bias[n] )
// Wave computes one 16x16 tile; 4 waves (distinct N tiles) per block.
// A: f16 [M][lda], W: f16 [Npad][ldw] (both K-padded, zero-filled).
// Fragment layouts per CDNA5 ISA 7.12.2 (16-bit A 16x32, B 32x16, f32 C 16x16).
// EPI: 0 = f32 store (+bias)        1 = f16 store (+bias)
//      2 = relu f16 (+bias)         3 = pooled = max(sigmoid(acc), aux0, aux1)
//      4 = dual store f32 + f16 (+bias)
// ---------------------------------------------------------------------------
template<int EPI>
__global__ __launch_bounds__(128) void gemm_wmma_kernel(
    const _Float16* __restrict__ A, int lda,
    const _Float16* __restrict__ W, int ldw,
    const float* __restrict__ bias,
    int K, int Nlog,
    float* __restrict__ outF, _Float16* __restrict__ outH,
    int ldo, int ncol_off,
    const _Float16* __restrict__ aux0, const _Float16* __restrict__ aux1,
    int ldaux)
{
    const int lane   = threadIdx.x & 31;
    const int wave   = threadIdx.x >> 5;
    const int mtile  = blockIdx.x;
    const int ntile  = blockIdx.y * 4 + wave;
    const int ntiles = (Nlog + 15) >> 4;
    if (ntile >= ntiles) return;

    const int m0 = mtile * 16;
    const int n0 = ntile * 16;
    const int nl = lane & 15;     // A row within tile AND B column within tile
    const int hi = lane >> 4;
    const int n  = n0 + nl;       // this lane's output column

    const _Float16* arow = A + (size_t)(m0 + nl) * lda;
    const _Float16* brow = W + (size_t)n * ldw;

    v8f acc = {};
    for (int k0 = 0; k0 < K; k0 += 32) {
        // A 16x32 f16 fragment: lanes 0-15 K{0..7,16..23}, lanes 16-31 K{8..15,24..31}
        v8h a0 = *(const v8h*)(arow + k0 + hi * 8);
        v8h a1 = *(const v8h*)(arow + k0 + 16 + hi * 8);
        v16h a;
        #pragma unroll
        for (int i = 0; i < 8; ++i) { a[i] = a0[i]; a[i + 8] = a1[i]; }
        // B 32x16 f16 fragment: same K order as A (lanes 0-15 K{0..7,16..23}, lanes 16-31 K{8..15,24..31})
        v8h b0 = *(const v8h*)(brow + k0 + hi * 8);
        v8h b1 = *(const v8h*)(brow + k0 + 16 + hi * 8);
        v16h bfr;
        #pragma unroll
        for (int i = 0; i < 8; ++i) { bfr[i] = b0[i]; bfr[i + 8] = b1[i]; }
        acc = __builtin_amdgcn_wmma_f32_16x16x32_f16(
            false, a, false, bfr, (short)0, acc, false, false);
    }

    if (n < Nlog) {
        const float bv = bias ? bias[n] : 0.f;
        const int mb = hi * 8;
        #pragma unroll
        for (int r = 0; r < 8; ++r) {
            const size_t m = (size_t)(m0 + mb + r);
            const float v = acc[r] + bv;
            if constexpr (EPI == 0) {
                outF[m * ldo + ncol_off + n] = v;
            } else if constexpr (EPI == 1) {
                outH[m * ldo + ncol_off + n] = (_Float16)v;
            } else if constexpr (EPI == 2) {
                outH[m * ldo + ncol_off + n] = (_Float16)fmaxf(v, 0.f);
            } else if constexpr (EPI == 3) {
                float s  = 1.f / (1.f + __expf(-acc[r]));   // no bias on this GEMM
                float p0 = (float)aux0[m * ldaux + n];
                float p1 = (float)aux1[m * ldaux + n];
                outH[m * ldo + n] = (_Float16)fmaxf(fmaxf(s, p0), p1);
            } else if constexpr (EPI == 4) {
                outF[m * ldo + n] = v;
                outH[m * ldo + n] = (_Float16)v;
            }
        }
    }
}

// ---------------------------------------------------------------------------
// Row l2-normalize: pnh = f16( pf / max(||pf||, eps) ), wave-per-row
// ---------------------------------------------------------------------------
__global__ __launch_bounds__(256) void rownorm_kernel(
    const float* __restrict__ pf, _Float16* __restrict__ pnh)
{
    const int lane = threadIdx.x & 31;
    const int b    = blockIdx.x * 8 + (threadIdx.x >> 5);
    const float* row = pf + (size_t)b * KP;
    float ss = 0.f;
    for (int k = lane; k < DFLAT; k += 32) { float v = row[k]; ss += v * v; }
    #pragma unroll
    for (int off = 16; off > 0; off >>= 1) ss += __shfl_xor(ss, off, 32);
    const float r = 1.f / fmaxf(sqrtf(ss), 1e-12f);
    for (int k = lane; k < DFLAT; k += 32)
        pnh[(size_t)b * KP + k] = (_Float16)(row[k] * r);
}

// ---------------------------------------------------------------------------
// Final: q = vf + adv - mean_actions(adv); softmax over atoms. In-place on out.
// ---------------------------------------------------------------------------
__global__ __launch_bounds__(256) void duel_softmax_kernel(
    float* __restrict__ out, const float* __restrict__ vff)
{
    __shared__ float s_adv[QROW];
    __shared__ float s_vf[ATOMS];
    __shared__ float s_mean[ATOMS];
    const int b = blockIdx.x, tid = threadIdx.x;
    float* row = out + (size_t)b * QROW;
    for (int i = tid; i < QROW; i += 256) s_adv[i] = row[i];
    if (tid < ATOMS) s_vf[tid] = vff[(size_t)b * 64 + tid];
    __syncthreads();
    if (tid < ATOMS) {
        float s = 0.f;
        for (int a = 0; a < NACT; ++a) s += s_adv[a * ATOMS + tid];
        s_mean[tid] = s * (1.f / NACT);
    }
    __syncthreads();
    if (tid < NACT) {
        float q[ATOMS];
        float mx = -3.4e38f;
        for (int j = 0; j < ATOMS; ++j) {
            q[j] = s_vf[j] + s_adv[tid * ATOMS + j] - s_mean[j];
            mx = fmaxf(mx, q[j]);
        }
        float sum = 0.f;
        for (int j = 0; j < ATOMS; ++j) { q[j] = __expf(q[j] - mx); sum += q[j]; }
        const float inv = 1.f / sum;
        for (int j = 0; j < ATOMS; ++j) row[tid * ATOMS + j] = q[j] * inv;
    }
}

// ---------------------------------------------------------------------------
extern "C" void kernel_launch(void* const* d_in, const int* in_sizes, int n_in,
                              void* d_out, int out_size, void* d_ws, size_t ws_size,
                              hipStream_t stream)
{
    (void)in_sizes; (void)n_in; (void)out_size; (void)ws_size;
    const float* in1  = (const float*)d_in[0];
    const float* in2  = (const float*)d_in[1];
    const float* ins  = (const float*)d_in[2];
    const float* W_g  = (const float*)d_in[3];
    const float* Wffl = (const float*)d_in[6];
    const float* bffl = (const float*)d_in[7];
    const float* Wp   = (const float*)d_in[12];
    const float* bp   = (const float*)d_in[13];
    const float* Wao  = (const float*)d_in[14];
    const float* bao  = (const float*)d_in[15];
    const float* Wa1  = (const float*)d_in[16];
    const float* ba1  = (const float*)d_in[17];
    const float* Wa2  = (const float*)d_in[18];
    const float* ba2  = (const float*)d_in[19];
    const float* Wv1  = (const float*)d_in[20];
    const float* bv1  = (const float*)d_in[21];
    const float* Wv2  = (const float*)d_in[22];
    const float* bv2  = (const float*)d_in[23];
    float* out = (float*)d_out;

    // -------- workspace layout (regions aliased across pipeline phases) -----
    char* ws = (char*)d_ws;
    _Float16* att1h   = (_Float16*)(ws + (  0ull << 20));  // 32 MiB
    _Float16* padsh   = (_Float16*)(ws + ( 32ull << 20));  // 32 MiB
    _Float16* pad2h   = (_Float16*)(ws + ( 64ull << 20));  // 32 MiB
    _Float16* pooledh = (_Float16*)(ws + ( 96ull << 20));  // 32 MiB
    float*    pf      = (float*)   (ws + (128ull << 20));  // 64 MiB
    float*    vff     = (float*)   (ws + (192ull << 20));  //  8 MiB
    char*     wbase   =             ws + (200ull << 20);
    // aliases (safe by stream ordering: producer of old region done first)
    _Float16* ph  = att1h;           // after GEMM1 consumed att1h
    _Float16* pnh = padsh;           // after GEMM1 consumed padsh
    _Float16* xh  = (_Float16*)pf;   // after rownorm consumed pf (B*544*2 <= 64M)
    _Float16* hah = pad2h;           // after GEMM1 consumed pad2h
    _Float16* hvh = pooledh;         // after GEMM2 consumed pooledh

    _Float16* Wg16   = (_Float16*)wbase;
    _Float16* Wp16   = Wg16   + 496 * 512;
    _Float16* Wffl16 = Wp16   + 496 * 512;
    _Float16* Wao16  = Wffl16 + 272 * 512;
    _Float16* Wa116  = Wao16  + 272 * 512;
    _Float16* Wa216  = Wa116  + 496 * 544;
    _Float16* Wv116  = Wa216  + 1536 * 512;
    _Float16* Wv216  = Wv116  + 496 * 544;

    auto conv = [&](_Float16* dst, const float* src, int N, int K,
                    int Npad, int Kpad, int trans) {
        int total = Npad * Kpad;
        convw_kernel<<<(total + 255) / 256, 256, 0, stream>>>(
            dst, src, N, K, Npad, Kpad, trans);
    };
    conv(Wg16,   W_g,  484,  484,  496, 512, 1);   // z = att @ W_g  (K-major src)
    conv(Wp16,   Wp,   484,  484,  496, 512, 0);
    conv(Wffl16, Wffl, 272,  484,  272, 512, 0);
    conv(Wao16,  Wao,  272,  484,  272, 512, 0);
    conv(Wa116,  Wa1,  484,  544,  496, 544, 0);
    conv(Wa216,  Wa2,  1530, 484, 1536, 512, 0);
    conv(Wv116,  Wv1,  484,  544,  496, 544, 0);
    conv(Wv216,  Wv2,  51,   484,   64, 512, 0);

    preprocess_kernel<<<BATCH / 8, 256, 0, stream>>>(
        in1, in2, ins, att1h, pad2h, padsh, pooledh);

    const int MT = BATCH / 16;  // 2048 M tiles
    auto gy = [](int nlog) { return (((nlog + 15) >> 4) + 3) / 4; };

    // GEMM1: pooled = max(sigmoid(att1 @ W_g), pads, pad2)   [fused epilogue]
    gemm_wmma_kernel<3><<<dim3(MT, gy(484)), 128, 0, stream>>>(
        att1h, KP, Wg16, KP, nullptr, KP, 484,
        nullptr, pooledh, KP, 0, padsh, pad2h, KP);

    // GEMM2: p = pooled @ Wp^T + bp  -> pf (f32) and ph (f16)
    gemm_wmma_kernel<4><<<dim3(MT, gy(484)), 128, 0, stream>>>(
        pooledh, KP, Wp16, KP, bp, KP, 484,
        pf, ph, KP, 0, nullptr, nullptr, 0);

    // l2norm(p) -> pnh
    rownorm_kernel<<<BATCH / 8, 256, 0, stream>>>(pf, pnh);

    // GEMM3a: mapped = l2norm(p) @ Wffl^T + bffl -> x[:, 0:272]
    gemm_wmma_kernel<1><<<dim3(MT, gy(272)), 128, 0, stream>>>(
        pnh, KP, Wffl16, KP, bffl, KP, 272,
        nullptr, xh, XDIM, 0, nullptr, nullptr, 0);

    // GEMM3b: att_out = p @ Wao^T + bao -> x[:, 272:544]
    gemm_wmma_kernel<1><<<dim3(MT, gy(272)), 128, 0, stream>>>(
        ph, KP, Wao16, KP, bao, KP, 272,
        nullptr, xh, XDIM, 272, nullptr, nullptr, 0);

    // GEMM4a: h_a = relu(x @ Wa1^T + ba1)
    gemm_wmma_kernel<2><<<dim3(MT, gy(484)), 128, 0, stream>>>(
        xh, XDIM, Wa116, XDIM, ba1, XDIM, 484,
        nullptr, hah, KP, 0, nullptr, nullptr, 0);

    // GEMM4b: h_v = relu(x @ Wv1^T + bv1)
    gemm_wmma_kernel<2><<<dim3(MT, gy(484)), 128, 0, stream>>>(
        xh, XDIM, Wv116, XDIM, bv1, XDIM, 484,
        nullptr, hvh, KP, 0, nullptr, nullptr, 0);

    // GEMM5: adv = h_a @ Wa2^T + ba2 -> staged directly into d_out (f32)
    gemm_wmma_kernel<0><<<dim3(MT, gy(1530)), 128, 0, stream>>>(
        hah, KP, Wa216, KP, ba2, KP, 1530,
        out, nullptr, QROW, 0, nullptr, nullptr, 0);

    // GEMM6: vf = h_v @ Wv2^T + bv2 -> vff (stride 64)
    gemm_wmma_kernel<0><<<dim3(MT, gy(51)), 128, 0, stream>>>(
        hvh, KP, Wv216, KP, bv2, KP, 51,
        vff, nullptr, 64, 0, nullptr, nullptr, 0);

    // Final: dueling combine + softmax over atoms, in-place on d_out
    duel_softmax_kernel<<<BATCH, 256, 0, stream>>>(out, vff);
}
